// EnhancedMultiHeadAttention_69793218560009
// MI455X (gfx1250) — matrix operand order repaired and checked by reading, not compiled, not run on hardware
//
#include <hip/hip_runtime.h>
#include <hip/hip_bf16.h>

typedef __attribute__((ext_vector_type(16))) _Float16 v16h;
typedef __attribute__((ext_vector_type(8)))  _Float16 v8h;
typedef __attribute__((ext_vector_type(8)))  float    v8f;
typedef __attribute__((ext_vector_type(4)))  unsigned u32x4;
typedef __attribute__((ext_vector_type(8)))  int      i32x8;
typedef __attribute__((ext_vector_type(4)))  int      i32x4;

#define SHUF16(lo, up) __builtin_shufflevector(lo, up, 0, 1, 2, 3, 4, 5, 6, 7, \
                                               8, 9, 10, 11, 12, 13, 14, 15)

// D = A(16x32 f16) * B(32x16 f16) + C(16x16 f32)
__device__ __forceinline__ v8f wmma_f16(v16h a, v16h b, v8f c) {
  return __builtin_amdgcn_wmma_f32_16x16x32_f16(false, a, false, b, (short)0, c, false, false);
}

// ---- Tensor Data Mover: 2D f16 tile (global -> LDS), row pad 8 DWORDs/row ----
// D# per CDNA5 ISA ch.8: group0 = {count, lds_addr, global_addr, type=2},
// group1 = {data_size=2B, pad cfg, tensor dims/strides, tile dims}.
// Row = tile_w*2 bytes = 64B (8x 8B units) -> pad_interval code 3, pad_amount
// code 7 (8 DWORDs = 32B) => LDS row stride 96B = 48 f16.
__device__ __forceinline__ void tdm_load_2d(unsigned lds_off, const void* gptr,
                                            unsigned tensor_w, unsigned tensor_h,
                                            unsigned stride_elems,
                                            unsigned tile_w, unsigned tile_h) {
  unsigned long long ga = (unsigned long long)(size_t)gptr;
  u32x4 g0;
  g0[0] = 1u;                                             // count=1, no gather
  g0[1] = lds_off;                                        // lds_addr (bytes)
  g0[2] = (unsigned)(ga & 0xffffffffu);                   // global_addr[31:0]
  g0[3] = (unsigned)((ga >> 32) & 0x01ffffffu) | (2u << 30);  // addr[56:32]|type=2
  i32x8 g1;
  g1[0] = (int)((1u << 16) |                              // data_size = 2 bytes
                (1u << 20) |                              // pad_enable
                (3u << 22) |                              // pad_interval: 8x8B
                (7u << 25));                              // pad_amount: 8 DWORDs
  g1[1] = (int)((tensor_w & 0xffffu) << 16);              // tensor_dim0[15:0]
  g1[2] = (int)(((tensor_w >> 16) & 0xffffu) | ((tensor_h & 0xffffu) << 16));
  g1[3] = (int)(((tensor_h >> 16) & 0xffffu) | ((tile_w & 0xffffu) << 16));
  g1[4] = (int)(tile_h & 0xffffu);                        // tile_dim1 (tile_dim2=0)
  g1[5] = (int)stride_elems;                              // dim0 stride [31:0]
  g1[6] = 0;
  g1[7] = 0;
  i32x4 z = {};
#if __clang_major__ >= 23
  i32x8 z8 = {};
  __builtin_amdgcn_tensor_load_to_lds(g0, g1, z, z, z8, 0);
#else
  __builtin_amdgcn_tensor_load_to_lds(g0, g1, z, z, 0);
#endif
}

// ---------------- fp32 -> fp16 convert ----------------
__global__ void cvt_f32_f16(const float* __restrict__ in, _Float16* __restrict__ out, int n) {
  int i = blockIdx.x * 256 + threadIdx.x;
  if (i < n) out[i] = (_Float16)in[i];
}

// 1024x1024 fp32 -> fp16 transpose (weights): out[n][k] = in[k][n]
__global__ void cvt_t_f32_f16(const float* __restrict__ in, _Float16* __restrict__ out) {
  int i = blockIdx.x * 256 + threadIdx.x;
  int k = i >> 10, n = i & 1023;
  out[(size_t)n * 1024 + k] = (_Float16)in[i];
}

// ---------------- tiled WMMA GEMM: (4096 x 1024) * (1024 x 1024)^T + bias ----
// A row-major [M,K]; BT is B transposed, row-major [N,K].
// Block tile 64(M) x 128(N), K-step 32; tiles staged by the TDM with row
// padding (LDS leading dim 48 f16); 8 waves each compute 32x32 (2x2 WMMA).
// mode 0: f16 out -> [B,H,L,64] head layout (scaled)
// mode 1: f32 out -> row-major [M,1024]
// mode 2: f16 out -> [B,H,64,L] transposed head layout (for V)
__global__ __launch_bounds__(256)
void gemm_wmma(const _Float16* __restrict__ A, const _Float16* __restrict__ BT,
               const float* __restrict__ bias, _Float16* __restrict__ out16,
               float* __restrict__ out32, int mode, float scale)
{
  constexpr int K = 1024, N = 1024, M = 4096;
  constexpr int LD = 48;                       // padded LDS leading dim (f16)
  __shared__ alignas(32) _Float16 As[64 * LD];
  __shared__ alignas(32) _Float16 Bs[128 * LD];

  const int tid  = threadIdx.x;
  const int m0   = blockIdx.y * 64;
  const int n0   = blockIdx.x * 128;
  const int w    = tid >> 5;
  const int lane = tid & 31;
  const int ln   = lane & 15;
  const int hi   = lane >> 4;
  const int wm   = (w & 1) * 32;
  const int wn   = (w >> 1) * 32;

  const unsigned asOff = (unsigned)(size_t)(void*)As;
  const unsigned bsOff = (unsigned)(size_t)(void*)Bs;

  v8f acc[2][2] = {};

  for (int k0 = 0; k0 < K; k0 += 32) {
    if (w == 0) {
      tdm_load_2d(asOff, A  + (size_t)m0 * K + k0, K, M, K, 32, 64);
      tdm_load_2d(bsOff, BT + (size_t)n0 * K + k0, K, N, K, 32, 128);
      __builtin_amdgcn_s_wait_tensorcnt(0);
    }
    __syncthreads();

    // A fragments: lane = M, k = hi*16 + e (contiguous 16 per lane-half)
    v16h af[2];
#pragma unroll
    for (int a = 0; a < 2; a++) {
      int row = wm + a * 16 + ln;
      v8h lo = *(const v8h*)(&As[row * LD + hi * 16]);
      v8h up = *(const v8h*)(&As[row * LD + hi * 16 + 8]);
      af[a] = SHUF16(lo, up);
    }
    // B fragments: lane = N, k = hi*16 + e -> one contiguous 32B LDS load
    v16h bf[2];
#pragma unroll
    for (int b = 0; b < 2; b++) {
      int col = wn + b * 16 + ln;
      bf[b] = *(const v16h*)(&Bs[col * LD + hi * 16]);
    }
#pragma unroll
    for (int a = 0; a < 2; a++)
#pragma unroll
      for (int b = 0; b < 2; b++)
        acc[a][b] = wmma_f16(af[a], bf[b], acc[a][b]);
    __syncthreads();
  }

  // epilogue: C/D layout: reg r -> M = r + 8*hi, lane -> N = ln
#pragma unroll
  for (int a = 0; a < 2; a++)
#pragma unroll
    for (int b = 0; b < 2; b++)
#pragma unroll
      for (int r = 0; r < 8; r++) {
        int gm = m0 + wm + a * 16 + r + 8 * hi;
        int gn = n0 + wn + b * 16 + ln;
        float v = (acc[a][b][r] + bias[gn]) * scale;
        int bq = gm >> 10, l = gm & 1023, h = gn >> 6, d = gn & 63;
        if (mode == 0) {
          out16[(((size_t)(bq * 16 + h)) * 1024 + l) * 64 + d] = (_Float16)v;
        } else if (mode == 1) {
          out32[(size_t)gm * N + gn] = v;
        } else {
          out16[(((size_t)(bq * 16 + h)) * 64 + d) * 1024 + l] = (_Float16)v;
        }
      }
}

// ---------------- flash-style attention with relative-position bias ---------
// grid.x = B*H*8 ; each block: 8 waves, each wave owns 16 query rows.
// qh/kh: [B*H, 1024, 64] f16 (kh pre-scaled by 1/sqrt(64)).
// vT: [B*H, 64, 1024] f16 (transposed). relh: [2047, 64] f16.
// ctxh out: [B, 1024, 1024] f16 (heads re-interleaved).
__global__ __launch_bounds__(256)
void attn_wmma(const _Float16* __restrict__ qh, const _Float16* __restrict__ kh,
               const _Float16* __restrict__ vT, const _Float16* __restrict__ relh,
               _Float16* __restrict__ ctxh)
{
  constexpr int LP = 40;                         // padded P-buffer stride (f16)
  __shared__ alignas(16) _Float16 pbuf[8 * 16 * LP];

  const int tid  = threadIdx.x;
  const int w    = tid >> 5;
  const int lane = tid & 31;
  const int ln   = lane & 15;
  const int hi   = lane >> 4;
  const int bh   = blockIdx.x >> 3;              // b*16 + h
  const int rblk = blockIdx.x & 7;
  const int i0   = rblk * 128 + w * 16;          // first query row of this wave

  const size_t ho = (size_t)bh * 1024 * 64;
  const _Float16* qb = qh + ho;
  const _Float16* kb = kh + ho;
  const _Float16* vb = vT + ho;
  _Float16* pp = &pbuf[w * 16 * LP];

  // load Q fragments once (A layout), K-dim 64 -> two k-steps of 32
  v16h qa[2];
#pragma unroll
  for (int s = 0; s < 2; s++) {
    const _Float16* qr = qb + (size_t)(i0 + ln) * 64 + s * 32 + hi * 16;
    v8h lo = *(const v8h*)(qr);
    v8h up = *(const v8h*)(qr + 8);
    qa[s] = SHUF16(lo, up);
  }

  v8f Oacc[4] = {};
  float mrow[8], lrow[8];
#pragma unroll
  for (int r = 0; r < 8; r++) { mrow[r] = -3.0e38f; lrow[r] = 0.f; }

  for (int j0 = 0; j0 < 1024; j0 += 32) {
    v8f sacc[2], g0[2], g1[2];
#pragma unroll
    for (int jt = 0; jt < 2; jt++) {
      int j = j0 + jt * 16;
      // S = q * k^T (k pre-scaled); contiguous B frag -> 32B vector load
      v8f s = {};
#pragma unroll
      for (int ss = 0; ss < 2; ss++) {
        v16h kf = *(const v16h*)(kb + (size_t)(j + ln) * 64 + ss * 32 + hi * 16);
        s = wmma_f16(qa[ss], kf, s);
      }
      sacc[jt] = s;
      // G = q * rel^T over the 31-row band; cols c = 0..30, c=31 zero pad
      int base = i0 - j + 1008;
      v8f g[2] = {{}, {}};
#pragma unroll
      for (int gc = 0; gc < 2; gc++) {
        int c = gc * 16 + ln;
#pragma unroll
        for (int ss = 0; ss < 2; ss++) {
          v16h rb = {};
          if (c < 31)
            rb = *(const v16h*)(relh + (size_t)(base + c) * 64 + ss * 32 + hi * 16);
          g[gc] = wmma_f16(qa[ss], rb, g[gc]);
        }
      }
      g0[jt] = g[0]; g1[jt] = g[1];
    }

    // bias extraction + online softmax (rows live in 16-lane halves)
    float pv[2][8];
#pragma unroll
    for (int r = 0; r < 8; r++) {
      int c    = r + 8 * hi - ln + 15;           // 0..30
      int srcl = hi * 16 + (c & 15);
      float rowmax = -3.0e38f;
#pragma unroll
      for (int jt = 0; jt < 2; jt++) {
        float b0 = __shfl(g0[jt][r], srcl, 32);
        float b1 = __shfl(g1[jt][r], srcl, 32);
        float bias = (c < 16) ? b0 : b1;
        float sv = fminf(fmaxf(sacc[jt][r], -100000.f), 100000.f) + bias;
        pv[jt][r] = sv;
        rowmax = fmaxf(rowmax, sv);
      }
#pragma unroll
      for (int off = 1; off < 16; off <<= 1)
        rowmax = fmaxf(rowmax, __shfl_xor(rowmax, off, 32));
      float nm   = fmaxf(mrow[r], rowmax);
      float corr = __expf(mrow[r] - nm);
      float rs = 0.f;
#pragma unroll
      for (int jt = 0; jt < 2; jt++) {
        float p = __expf(pv[jt][r] - nm);
        pv[jt][r] = p;
        rs += p;
      }
#pragma unroll
      for (int off = 1; off < 16; off <<= 1)
        rs += __shfl_xor(rs, off, 32);
      lrow[r] = lrow[r] * corr + rs;
      mrow[r] = nm;
#pragma unroll
      for (int t = 0; t < 4; t++) Oacc[t][r] *= corr;
    }

    // route P (C layout) -> A layout through wave-private LDS; only an LDS
    // counter wait is needed (no cross-wave sharing -> no barrier)
#pragma unroll
    for (int jt = 0; jt < 2; jt++)
#pragma unroll
      for (int r = 0; r < 8; r++)
        pp[(r + 8 * hi) * LP + jt * 16 + ln] = (_Float16)pv[jt][r];
    asm volatile("s_wait_dscnt 0" ::: "memory");

    v8h plo = *(const v8h*)(&pp[ln * LP + hi * 16]);
    v8h pup = *(const v8h*)(&pp[ln * LP + hi * 16 + 8]);
    v16h pa = SHUF16(plo, pup);

    // O += P * V  (V transposed -> contiguous 32B B-frag loads)
#pragma unroll
    for (int t = 0; t < 4; t++) {
      v16h vf = *(const v16h*)(vb + (size_t)(t * 16 + ln) * 1024 + j0 + hi * 16);
      Oacc[t] = wmma_f16(pa, vf, Oacc[t]);
    }
  }

  // epilogue: ctx = O / l, stored f16 in [B, L, HID] with heads re-interleaved
  const int b = bh >> 4, h = bh & 15;
#pragma unroll
  for (int t = 0; t < 4; t++)
#pragma unroll
    for (int r = 0; r < 8; r++) {
      int m = r + 8 * hi;
      float v = Oacc[t][r] / lrow[r];
      ctxh[((size_t)(b * 1024 + (i0 + m))) * 1024 + h * 64 + t * 16 + ln] = (_Float16)v;
    }
}

// ---------------------------------------------------------------------------
extern "C" void kernel_launch(void* const* d_in, const int* in_sizes, int n_in,
                              void* d_out, int out_size, void* d_ws, size_t ws_size,
                              hipStream_t stream) {
  (void)in_sizes; (void)n_in; (void)out_size; (void)ws_size;
  const float* x   = (const float*)d_in[0];
  const float* Wq  = (const float*)d_in[1];
  const float* bq  = (const float*)d_in[2];
  const float* Wk  = (const float*)d_in[3];
  const float* bk  = (const float*)d_in[4];
  const float* Wv  = (const float*)d_in[5];
  const float* bv  = (const float*)d_in[6];
  const float* Wo  = (const float*)d_in[7];
  const float* bo  = (const float*)d_in[8];
  const float* rel = (const float*)d_in[9];
  float* out = (float*)d_out;

  char* ws = (char*)d_ws;
  size_t off = 0;
  auto take = [&](size_t bytes) -> char* {
    char* p = ws + off; off += (bytes + 255) & ~(size_t)255; return p;
  };
  _Float16* xh   = (_Float16*)take(4096ull * 1024 * 2);   // x in f16
  _Float16* WqT  = (_Float16*)take(1024ull * 1024 * 2);   // W^T in f16
  _Float16* WkT  = (_Float16*)take(1024ull * 1024 * 2);
  _Float16* WvT  = (_Float16*)take(1024ull * 1024 * 2);
  _Float16* WoT  = (_Float16*)take(1024ull * 1024 * 2);
  _Float16* relh = (_Float16*)take(2047ull * 64 * 2);
  _Float16* qh   = (_Float16*)take(64ull * 1024 * 64 * 2); // [B*H, L, 64]
  _Float16* kh   = (_Float16*)take(64ull * 1024 * 64 * 2); // [B*H, L, 64]
  _Float16* vTh  = (_Float16*)take(64ull * 1024 * 64 * 2); // [B*H, 64, L]
  _Float16* ctxh = (_Float16*)take(4096ull * 1024 * 2);    // [B, L, HID]

  // fp32 -> fp16 conversions (weights transposed for TDM-friendly B tiles)
  cvt_f32_f16<<<(4194304 + 255) / 256, 256, 0, stream>>>(x, xh, 4194304);
  cvt_t_f32_f16<<<4096, 256, 0, stream>>>(Wq, WqT);
  cvt_t_f32_f16<<<4096, 256, 0, stream>>>(Wk, WkT);
  cvt_t_f32_f16<<<4096, 256, 0, stream>>>(Wv, WvT);
  cvt_t_f32_f16<<<4096, 256, 0, stream>>>(Wo, WoT);
  cvt_f32_f16<<<(131008 + 255) / 256, 256, 0, stream>>>(rel, relh, 131008);

  // projections: q,k,v  (k scaled by 1/sqrt(64); v stored transposed)
  dim3 ggrid(1024 / 128, 4096 / 64);
  gemm_wmma<<<ggrid, 256, 0, stream>>>(xh, WqT, bq, qh, nullptr, 0, 1.0f);
  gemm_wmma<<<ggrid, 256, 0, stream>>>(xh, WkT, bk, kh, nullptr, 0, 0.125f);
  gemm_wmma<<<ggrid, 256, 0, stream>>>(xh, WvT, bv, vTh, nullptr, 2, 1.0f);

  // attention with relative bias
  attn_wmma<<<4 * 16 * 8, 256, 0, stream>>>(qh, kh, vTh, relh, ctxh);

  // output projection -> f32
  gemm_wmma<<<ggrid, 256, 0, stream>>>(ctxh, WoT, bo, nullptr, out, 1, 1.0f);
}
